// VLA_Adapter_L1RegressionActionHead_6975026888970
// MI455X (gfx1250) — matrix operand order repaired and checked by reading, not compiled, not run on hardware
//
#include <hip/hip_runtime.h>
#include <math.h>

// ---------------- Types / constants ----------------
typedef __bf16 bf16;
typedef __attribute__((ext_vector_type(16))) __bf16 v16bf;
typedef __attribute__((ext_vector_type(8)))  float  v8f;

#define B_    2
#define T_    2048
#define DIN_  2048
#define D_    512
#define H_    8
#define HD_   64
#define R_    (B_ * T_)      // 4096 rows
#define NB_   2

// ---------------- WMMA helpers ----------------
__device__ inline v8f wmma_bf16(v16bf a, v16bf b, v8f c) {
  // (neg_a, A, neg_b, B, c_mod, C, reuse_a, reuse_b)
  return __builtin_amdgcn_wmma_f32_16x16x32_bf16(false, a, false, b, (short)0, c, false, false);
}

// A-matrix 16x32 bf16 fragment from row-major storage (stride in elements).
// ISA: lanes 0-15: row m=lane, K = k0+0..7 (elems 0..7), k0+16..23 (elems 8..15)
//      lanes 16-31: row m=lane-16, K = k0+8..15, k0+24..31
__device__ inline v16bf load_frag_a(const bf16* base, int stride, int k0) {
  int lane = threadIdx.x & 31;
  int m    = lane & 15;
  int ko   = k0 + ((lane & 16) ? 8 : 0);
  const bf16* pp = base + (size_t)m * stride + ko;
  v16bf r;
#pragma unroll
  for (int i = 0; i < 8; i++) r[i] = pp[i];
#pragma unroll
  for (int i = 0; i < 8; i++) r[i + 8] = pp[16 + i];
  return r;
}

// B-matrix 32x16 bf16 fragment from N-major storage (row n holds K-contiguous data).
// B striping (matches A): lanes 0-15 hold K=k0+0..7,k0+16..23 for column n=lane; lanes 16-31 hold K=k0+8..15,k0+24..31.
__device__ inline v16bf load_frag_b(const bf16* baseNmajor, int stride, int k0) {
  int lane = threadIdx.x & 31;
  int n    = lane & 15;
  int ko   = k0 + ((lane & 16) ? 8 : 0);
  const bf16* pp = baseNmajor + (size_t)n * stride + ko;
  v16bf r;
#pragma unroll
  for (int i = 0; i < 8; i++) r[i] = pp[i];
#pragma unroll
  for (int i = 0; i < 8; i++) r[i + 8] = pp[16 + i];
  return r;
}

// A fragment from a 16 x stride row-major LDS tile at K offset k0.
__device__ inline v16bf load_frag_a_lds(const bf16* pb, int stride, int k0) {
  int lane = threadIdx.x & 31;
  int m    = lane & 15;
  int ko   = k0 + ((lane & 16) ? 8 : 0);
  const bf16* pp = pb + m * stride + ko;
  v16bf r;
#pragma unroll
  for (int i = 0; i < 8; i++) r[i] = pp[i];
#pragma unroll
  for (int i = 0; i < 8; i++) r[i + 8] = pp[16 + i];
  return r;
}

__device__ inline float redmax16(float v) {
#pragma unroll
  for (int m = 1; m < 16; m <<= 1) v = fmaxf(v, __shfl_xor(v, m, 32));
  return v;
}
__device__ inline float redsum16(float v) {
#pragma unroll
  for (int m = 1; m < 16; m <<= 1) v += __shfl_xor(v, m, 32);
  return v;
}

// ---------------- LayerNorm: fp32 in -> bf16 out ----------------
__global__ __launch_bounds__(256) void ln_kernel(const float* __restrict__ x,
                                                 const float* __restrict__ g,
                                                 const float* __restrict__ b,
                                                 bf16* __restrict__ out, int C) {
  __shared__ float ssum[8], ssq[8];
  int row = blockIdx.x;
  const float* xr = x + (size_t)row * C;
  float s = 0.f, q = 0.f;
  for (int c = threadIdx.x; c < C; c += blockDim.x) { float v = xr[c]; s += v; q += v * v; }
#pragma unroll
  for (int m = 1; m < 32; m <<= 1) { s += __shfl_xor(s, m, 32); q += __shfl_xor(q, m, 32); }
  if ((threadIdx.x & 31) == 0) { ssum[threadIdx.x >> 5] = s; ssq[threadIdx.x >> 5] = q; }
  __syncthreads();
  if (threadIdx.x == 0) {
    float S = 0.f, Q = 0.f;
#pragma unroll
    for (int i = 0; i < 8; i++) { S += ssum[i]; Q += ssq[i]; }
    ssum[0] = S; ssq[0] = Q;
  }
  __syncthreads();
  float mean = ssum[0] / (float)C;
  float var  = ssq[0] / (float)C - mean * mean;
  float rstd = rsqrtf(var + 1e-5f);
  bf16* orow = out + (size_t)row * C;
  for (int c = threadIdx.x; c < C; c += blockDim.x)
    orow[c] = (bf16)((xr[c] - mean) * rstd * g[c] + b[c]);
}

// ---------------- Weight transpose+convert: W(K,N) f32 -> Wt(N,K) bf16 ----------------
__global__ __launch_bounds__(256) void wtrans_kernel(const float* __restrict__ W,
                                                     bf16* __restrict__ Wt, int K, int N) {
  size_t i = (size_t)blockIdx.x * blockDim.x + threadIdx.x;
  if (i >= (size_t)K * N) return;
  int k = (int)(i / N), n = (int)(i % N);
  Wt[(size_t)n * K + k] = (bf16)W[(size_t)k * N + n];
}

// ---------------- GEMM: (MT*16)x(NT*16) wave tile ----------------
// C = relu?(A(MxK) * Wt(NxK)^T + bias + resid); per K=32 step: MT+NT frag loads, MT*NT WMMAs.
template <int MT, int NT>
__global__ __launch_bounds__(256) void gemm_bf16_kernel(
    const bf16* __restrict__ A, const bf16* __restrict__ Wt,
    const float* __restrict__ bias, const float* __restrict__ resid,
    float* __restrict__ outF, bf16* __restrict__ outB,
    int M, int N, int K, int relu) {
  int wave   = blockIdx.x * 8 + (threadIdx.x >> 5);
  int ntiles = N / (16 * NT);
  int mtiles = M / (16 * MT);
  if (wave >= mtiles * ntiles) return;
  int tm = wave / ntiles, tn = wave - tm * ntiles;

  v8f c[MT][NT];
#pragma unroll
  for (int mt = 0; mt < MT; mt++)
#pragma unroll
    for (int t = 0; t < NT; t++) c[mt][t] = (v8f){0.f,0.f,0.f,0.f,0.f,0.f,0.f,0.f};

  for (int k = 0; k < K; k += 32) {
    if (k + 32 < K)
      __builtin_prefetch(A + (size_t)((tm * MT) << 4) * K + (size_t)(threadIdx.x & 15) * K + k + 32, 0, 1);
    v16bf a[MT];
#pragma unroll
    for (int mt = 0; mt < MT; mt++)
      a[mt] = load_frag_a(A + (size_t)((tm * MT + mt) << 4) * K, K, k);
#pragma unroll
    for (int t = 0; t < NT; t++) {
      v16bf bfr = load_frag_b(Wt + (size_t)((tn * NT + t) << 4) * K, K, k);
#pragma unroll
      for (int mt = 0; mt < MT; mt++) c[mt][t] = wmma_bf16(a[mt], bfr, c[mt][t]);
    }
  }

  int lane = threadIdx.x & 31;
  int nl = lane & 15, mo = (lane & 16) ? 8 : 0;
#pragma unroll
  for (int t = 0; t < NT; t++) {
    int gn = ((tn * NT + t) << 4) + nl;
    float bv = bias ? bias[gn] : 0.f;
#pragma unroll
    for (int mt = 0; mt < MT; mt++) {
#pragma unroll
      for (int i = 0; i < 8; i++) {
        int gm = ((tm * MT + mt) << 4) + mo + i;
        size_t o = (size_t)gm * N + gn;
        float v = c[mt][t][i] + bv;
        if (resid) v += resid[o];
        if (relu) v = fmaxf(v, 0.f);
        if (outF) outF[o] = v;
        if (outB) outB[o] = (bf16)v;
      }
    }
  }
}

// ---------------- RoPE (reference semantics) fp32 -> bf16, rows of length D_ ----------
__global__ __launch_bounds__(256) void rope_kernel(const float* __restrict__ in,
                                                   bf16* __restrict__ out,
                                                   int total, int posmod) {
  int idx = blockIdx.x * blockDim.x + threadIdx.x;
  if (idx >= total) return;
  int row = idx / D_;
  int d   = idx - row * D_;
  int dh  = d & (HD_ - 1);
  int pos = row % posmod;
  float invf = __powf(10000.f, -(float)(dh & 31) / 32.f);
  float ang  = (float)pos * invf;
  float cv = __cosf(ang), sv = __sinf(ang);
  float v  = in[idx];
  float pr = (dh & 1) ? in[idx - 1] : -in[idx + 1];
  out[idx] = (bf16)(v * cv + pr * sv);
}

// ---------------- fp32 -> bf16 ----------------
__global__ __launch_bounds__(256) void cvt_kernel(const float* __restrict__ in,
                                                  bf16* __restrict__ out, int n) {
  int i = blockIdx.x * blockDim.x + threadIdx.x;
  if (i < n) out[i] = (bf16)in[i];
}

// ---------------- V transpose: vs(B,T,D) bf16 -> vT(B,H,HD,T) bf16 ----------------
__global__ __launch_bounds__(256) void vtrans_kernel(const bf16* __restrict__ vs,
                                                     bf16* __restrict__ vT) {
  size_t i = (size_t)blockIdx.x * blockDim.x + threadIdx.x;
  if (i >= (size_t)R_ * D_) return;
  int d = (int)(i % D_);
  int s = (int)((i / D_) % T_);
  int b = (int)(i / ((size_t)D_ * T_));
  int h = d >> 6, dh = d & 63;
  vT[((size_t)((b * H_ + h) * HD_ + dh)) * T_ + s] = vs[i];
}

// ---------------- Adapter/task projections (tiny) ----------------
// r: 0..3 ka (b,tok), 4..7 va, 8..9 kt, 10..11 vt
__global__ __launch_bounds__(256) void small_proj_kernel(
    const float* __restrict__ h_a, const float* __restrict__ h_t, const float* __restrict__ p,
    const float* __restrict__ kaW, const float* __restrict__ kab,
    const float* __restrict__ vaW, const float* __restrict__ vab,
    const float* __restrict__ ktW, const float* __restrict__ ktb,
    const float* __restrict__ vtW, const float* __restrict__ vtb,
    int idx, float* __restrict__ kaO, float* __restrict__ vaO,
    float* __restrict__ ktO, float* __restrict__ vtO) {
  int tid = blockIdx.x * blockDim.x + threadIdx.x;
  if (tid >= 12 * D_) return;
  int n = tid & (D_ - 1);
  int r = tid >> 9;
  const float* vin; const float* W; const float* bb; float* o;
  if (r < 4) {
    int b = r >> 1, tok = r & 1;
    vin = tok ? (p + (size_t)b * D_) : (h_a + ((size_t)b * (NB_ + 1) + idx) * D_);
    W = kaW; bb = kab; o = kaO + (size_t)(b * 2 + tok) * D_;
  } else if (r < 8) {
    int rr = r - 4, b = rr >> 1, tok = rr & 1;
    vin = tok ? (p + (size_t)b * D_) : (h_a + ((size_t)b * (NB_ + 1) + idx) * D_);
    W = vaW; bb = vab; o = vaO + (size_t)(b * 2 + tok) * D_;
  } else if (r < 10) {
    int b = r - 8;
    vin = h_t + ((size_t)b * (NB_ + 1) + idx) * D_;
    W = ktW; bb = ktb; o = ktO + (size_t)b * D_;
  } else {
    int b = r - 10;
    vin = h_t + ((size_t)b * (NB_ + 1) + idx) * D_;
    W = vtW; bb = vtb; o = vtO + (size_t)b * D_;
  }
  float s = bb[n];
  for (int k = 0; k < D_; k++) s += vin[k] * W[(size_t)k * D_ + n];
  o[n] = s;
}

// ---------------- Flash attention: 32 q-rows/wave, 64-key softmax blocks ------------
// Per 64-key step: 8 K-frag + 8 V-frag loads feed 32 WMMAs; one max-reduce + one
// sum-reduce per row (cross-lane butterflies amortized over 64 keys).
__global__ __launch_bounds__(128) void attn_kernel(
    const bf16* __restrict__ q,  const bf16* __restrict__ ks, const bf16* __restrict__ vT,
    const bf16* __restrict__ ka, const bf16* __restrict__ va,
    const bf16* __restrict__ kt, const bf16* __restrict__ vt,
    const float* __restrict__ gptr, bf16* __restrict__ out) {
  __shared__ bf16  pbuf[4][2][16 * 64];
  __shared__ float sext[4][32][3];

  int w    = threadIdx.x >> 5;
  int lane = threadIdx.x & 31;
  int gw   = blockIdx.x * 4 + w;        // 0..1023
  int trow = gw & 63;                    // T_/32
  int h    = (gw >> 6) & (H_ - 1);
  int b    = gw >> 9;
  int t0   = trow << 5;
  const float scale = 0.125f;            // 1/sqrt(HD)
  float tanhg = tanhf(gptr[0]);

  v16bf a[2][2];
#pragma unroll
  for (int rs = 0; rs < 2; rs++) {
    const bf16* qb_ = q + ((size_t)(b * T_ + t0 + rs * 16)) * D_ + h * HD_;
    a[rs][0] = load_frag_a(qb_, D_, 0);
    a[rs][1] = load_frag_a(qb_, D_, 32);
  }

  float mrow[2][8], lrow[2][8];
  v8f acc[2][4];
#pragma unroll
  for (int rs = 0; rs < 2; rs++) {
#pragma unroll
    for (int i = 0; i < 8; i++) { mrow[rs][i] = -1e30f; lrow[rs][i] = 0.f; }
#pragma unroll
    for (int t = 0; t < 4; t++) acc[rs][t] = (v8f){0.f,0.f,0.f,0.f,0.f,0.f,0.f,0.f};
  }

  int nl = lane & 15, mo = (lane & 16) ? 8 : 0;
  const bf16* vbase = vT + ((size_t)((b * H_ + h) * HD_)) * T_;

  for (int j = 0; j < T_; j += 64) {
    // 4 key sub-tiles x 2 K-chunks, shared across both row-sets
    const bf16* kb = ks + ((size_t)(b * T_ + j)) * D_ + h * HD_;
    v16bf kf[4][2];
#pragma unroll
    for (int s = 0; s < 4; s++) {
      kf[s][0] = load_frag_b(kb + (size_t)(s * 16) * D_, D_, 0);
      kf[s][1] = load_frag_b(kb + (size_t)(s * 16) * D_, D_, 32);
    }

#pragma unroll
    for (int rs = 0; rs < 2; rs++) {
      v8f sv[4];
#pragma unroll
      for (int s = 0; s < 4; s++) {
        sv[s] = (v8f){0.f,0.f,0.f,0.f,0.f,0.f,0.f,0.f};
        sv[s] = wmma_bf16(a[rs][0], kf[s][0], sv[s]);
        sv[s] = wmma_bf16(a[rs][1], kf[s][1], sv[s]);
      }
      bf16* pb = pbuf[w][rs];
#pragma unroll
      for (int i = 0; i < 8; i++) {
        float e[4];
#pragma unroll
        for (int s = 0; s < 4; s++) e[s] = sv[s][i] * scale;
        float lm   = fmaxf(fmaxf(e[0], e[1]), fmaxf(e[2], e[3]));
        float tmax = redmax16(lm);
        float nm   = fmaxf(mrow[rs][i], tmax);
        float al   = __expf(mrow[rs][i] - nm);
        float pv[4]; float ps = 0.f;
#pragma unroll
        for (int s = 0; s < 4; s++) { pv[s] = __expf(e[s] - nm); ps += pv[s]; }
        float rsum = redsum16(ps);
        lrow[rs][i] = lrow[rs][i] * al + rsum;
        mrow[rs][i] = nm;
#pragma unroll
        for (int t = 0; t < 4; t++) acc[rs][t][i] *= al;
#pragma unroll
        for (int s = 0; s < 4; s++) pb[(mo + i) * 64 + s * 16 + nl] = (bf16)pv[s];
      }
    }
    __syncthreads();

    v16bf pa[2][2];
#pragma unroll
    for (int rs = 0; rs < 2; rs++) {
      pa[rs][0] = load_frag_a_lds(pbuf[w][rs], 64, 0);
      pa[rs][1] = load_frag_a_lds(pbuf[w][rs], 64, 32);
    }
#pragma unroll
    for (int t = 0; t < 4; t++) {
      const bf16* vrow = vbase + (size_t)(t * 16) * T_ + j;
      v16bf vb0 = load_frag_b(vrow, T_, 0);
      v16bf vb1 = load_frag_b(vrow, T_, 32);
      acc[0][t] = wmma_bf16(pa[0][0], vb0, acc[0][t]);
      acc[0][t] = wmma_bf16(pa[0][1], vb1, acc[0][t]);
      acc[1][t] = wmma_bf16(pa[1][0], vb0, acc[1][t]);
      acc[1][t] = wmma_bf16(pa[1][1], vb1, acc[1][t]);
    }
    __syncthreads();
  }

  // Extra columns: 2 adapter keys (RoPE'd) + task key * tanh(g). One row per lane.
  {
    const bf16* qr  = q  + ((size_t)(b * T_ + t0 + lane)) * D_ + h * HD_;
    const bf16* k0p = ka + ((size_t)(b * 2 + 0)) * D_ + h * HD_;
    const bf16* k1p = ka + ((size_t)(b * 2 + 1)) * D_ + h * HD_;
    const bf16* k2p = kt + ((size_t)b) * D_ + h * HD_;
    float e0 = 0.f, e1 = 0.f, e2 = 0.f;
    for (int d = 0; d < HD_; d++) {
      float qv = (float)qr[d];
      e0 += qv * (float)k0p[d];
      e1 += qv * (float)k1p[d];
      e2 += qv * (float)k2p[d];
    }
    sext[w][lane][0] = e0 * scale;
    sext[w][lane][1] = e1 * scale;
    sext[w][lane][2] = e2 * scale * tanhg;
  }
  __syncthreads();

  float ve0[4], ve1[4], ve2[4];
#pragma unroll
  for (int t = 0; t < 4; t++) {
    int d = t * 16 + nl;
    ve0[t] = (float)va[((size_t)(b * 2 + 0)) * D_ + h * HD_ + d];
    ve1[t] = (float)va[((size_t)(b * 2 + 1)) * D_ + h * HD_ + d];
    ve2[t] = (float)vt[((size_t)b) * D_ + h * HD_ + d];
  }
#pragma unroll
  for (int rs = 0; rs < 2; rs++) {
#pragma unroll
    for (int i = 0; i < 8; i++) {
      int r = rs * 16 + mo + i;
      float e0 = sext[w][r][0], e1 = sext[w][r][1], e2 = sext[w][r][2];
      float nm = fmaxf(fmaxf(mrow[rs][i], e0), fmaxf(e1, e2));
      float al = __expf(mrow[rs][i] - nm);
      float x0 = __expf(e0 - nm), x1 = __expf(e1 - nm), x2 = __expf(e2 - nm);
      float L  = lrow[rs][i] * al + x0 + x1 + x2;
      float iv = 1.f / L;
#pragma unroll
      for (int t = 0; t < 4; t++) {
        float o = (acc[rs][t][i] * al + x0 * ve0[t] + x1 * ve1[t] + x2 * ve2[t]) * iv;
        out[((size_t)(b * T_ + t0 + r)) * D_ + h * HD_ + t * 16 + nl] = (bf16)o;
      }
    }
  }
}

// ---------------- Host-side orchestration ----------------
extern "C" void kernel_launch(void* const* d_in, const int* in_sizes, int n_in,
                              void* d_out, int out_size, void* d_ws, size_t ws_size,
                              hipStream_t stream) {
  const float* x    = (const float*)d_in[0];
  const float* h_a  = (const float*)d_in[1];
  const float* h_t  = (const float*)d_in[2];
  const float* p    = (const float*)d_in[3];
  const float* ln1g = (const float*)d_in[4];
  const float* ln1b = (const float*)d_in[5];
  const float* fc1W = (const float*)d_in[6];
  const float* fc1b = (const float*)d_in[7];
  const float* qW   = (const float*)d_in[8];
  const float* qb   = (const float*)d_in[9];
  const float* ksW  = (const float*)d_in[10];
  const float* ksb  = (const float*)d_in[11];
  const float* vsW  = (const float*)d_in[12];
  const float* vsb  = (const float*)d_in[13];
  const float* kaW  = (const float*)d_in[14];
  const float* kab  = (const float*)d_in[15];
  const float* vaW  = (const float*)d_in[16];
  const float* vab  = (const float*)d_in[17];
  const float* ktW  = (const float*)d_in[18];
  const float* ktb  = (const float*)d_in[19];
  const float* vtW  = (const float*)d_in[20];
  const float* vtb  = (const float*)d_in[21];
  const float* oW   = (const float*)d_in[22];
  const float* ob   = (const float*)d_in[23];
  const float* gg   = (const float*)d_in[24];
  const float* lnG  = (const float*)d_in[25];
  const float* lnB  = (const float*)d_in[26];
  const float* fW   = (const float*)d_in[27];
  const float* fb   = (const float*)d_in[28];
  const float* ln2g = (const float*)d_in[29];
  const float* ln2b = (const float*)d_in[30];
  const float* fc2W = (const float*)d_in[31];
  const float* fc2b = (const float*)d_in[32];

  char* ws = (char*)d_ws;
  const size_t MB = 1u << 20;
  // Region A (16MB): xn_bf for fc1, then reused for q/ks/vs/vT bf16
  bf16*  xn_bf  = (bf16*)(ws + 0);
  bf16*  q_bf   = (bf16*)(ws + 0);
  bf16*  ks_bf  = (bf16*)(ws + 4  * MB);
  bf16*  vs_bf  = (bf16*)(ws + 8  * MB);
  bf16*  vT_bf  = (bf16*)(ws + 12 * MB);
  float* stage  = (float*)(ws + 16 * MB);   // 8MB: q/ks pre-RoPE f32, then y f32
  float* xf     = (float*)(ws + 24 * MB);   // 8MB: current activation f32 (residual)
  bf16*  xb     = (bf16*)(ws + 32 * MB);    // 4MB: current activation bf16
  bf16*  attnbf = (bf16*)(ws + 36 * MB);    // 4MB
  bf16*  ylnbf  = (bf16*)(ws + 40 * MB);    // 4MB
  bf16*  wt_big = (bf16*)(ws + 44 * MB);    // 2MB (512x2048)
  bf16*  wt_sm  = (bf16*)(ws + 46 * MB);    // 512KB (512x512)
  char*  sm     = ws + 46 * MB + (512u << 10);
  float* ka_f = (float*)sm; sm += 8192;
  float* va_f = (float*)sm; sm += 8192;
  float* kt_f = (float*)sm; sm += 4096;
  float* vt_f = (float*)sm; sm += 4096;
  bf16*  ka_b = (bf16*)sm;  sm += 4096;
  bf16*  va_b = (bf16*)sm;  sm += 4096;
  bf16*  kt_b = (bf16*)sm;  sm += 2048;
  bf16*  vt_b = (bf16*)sm;  sm += 2048;

  // Big GEMMs: 32x64 wave tiles. fc2 head: 32x32 wave tiles (N=32).
  auto gemm = [&](const bf16* A, const bf16* Wt, const float* bias, const float* res,
                  float* oF, bf16* oB, int M, int N, int K, int relu) {
    int waves = (M / 32) * (N / 64);
    gemm_bf16_kernel<2, 4><<<(waves + 7) / 8, 256, 0, stream>>>(A, Wt, bias, res, oF, oB, M, N, K, relu);
  };

  // ---- Stem: LN1 -> fc1 + relu ----
  ln_kernel<<<R_, 256, 0, stream>>>(x, ln1g, ln1b, xn_bf, DIN_);
  wtrans_kernel<<<(DIN_ * D_) / 256, 256, 0, stream>>>(fc1W, wt_big, DIN_, D_);
  gemm(xn_bf, wt_big, fc1b, nullptr, xf, xb, R_, D_, DIN_, 1);

  for (int blk = 0; blk < NB_; blk++) {
    size_t wo = (size_t)blk * D_ * D_;
    int bo2 = blk * D_;
    // q projection + RoPE
    wtrans_kernel<<<(D_ * D_) / 256, 256, 0, stream>>>(qW + wo, wt_sm, D_, D_);
    gemm(xb, wt_sm, qb + bo2, nullptr, stage, nullptr, R_, D_, D_, 0);
    rope_kernel<<<(R_ * D_) / 256, 256, 0, stream>>>(stage, q_bf, R_ * D_, T_);
    // ks projection + RoPE
    wtrans_kernel<<<(D_ * D_) / 256, 256, 0, stream>>>(ksW + wo, wt_sm, D_, D_);
    gemm(xb, wt_sm, ksb + bo2, nullptr, stage, nullptr, R_, D_, D_, 0);
    rope_kernel<<<(R_ * D_) / 256, 256, 0, stream>>>(stage, ks_bf, R_ * D_, T_);
    // vs projection (bf16 direct) + transpose to (B,H,HD,T)
    wtrans_kernel<<<(D_ * D_) / 256, 256, 0, stream>>>(vsW + wo, wt_sm, D_, D_);
    gemm(xb, wt_sm, vsb + bo2, nullptr, nullptr, vs_bf, R_, D_, D_, 0);
    vtrans_kernel<<<(R_ * D_) / 256, 256, 0, stream>>>(vs_bf, vT_bf);
    // adapter/task K/V (tiny), RoPE ka, convert others
    small_proj_kernel<<<(12 * D_) / 256, 256, 0, stream>>>(
        h_a, h_t, p, kaW + wo, kab + bo2, vaW + wo, vab + bo2,
        ktW + wo, ktb + bo2, vtW + wo, vtb + bo2, blk + 1, ka_f, va_f, kt_f, vt_f);
    rope_kernel<<<(4 * D_) / 256, 256, 0, stream>>>(ka_f, ka_b, 4 * D_, 2);
    cvt_kernel<<<(4 * D_) / 256, 256, 0, stream>>>(va_f, va_b, 4 * D_);
    cvt_kernel<<<(2 * D_) / 256, 256, 0, stream>>>(kt_f, kt_b, 2 * D_);
    cvt_kernel<<<(2 * D_) / 256, 256, 0, stream>>>(vt_f, vt_b, 2 * D_);
    // flash attention: 1024 waves, 4 waves/block
    attn_kernel<<<(B_ * H_ * (T_ / 32)) / 4, 128, 0, stream>>>(
        q_bf, ks_bf, vT_bf, ka_b, va_b, kt_b, vt_b, gg + blk, attnbf);
    // output projection + residual -> y (f32 in stage)
    wtrans_kernel<<<(D_ * D_) / 256, 256, 0, stream>>>(oW + wo, wt_sm, D_, D_);
    gemm(attnbf, wt_sm, ob + bo2, xf, stage, nullptr, R_, D_, D_, 0);
    // LN -> FFN + relu -> next activation
    ln_kernel<<<R_, 256, 0, stream>>>(stage, lnG + bo2, lnB + bo2, ylnbf, D_);
    wtrans_kernel<<<(D_ * D_) / 256, 256, 0, stream>>>(fW + wo, wt_sm, D_, D_);
    gemm(ylnbf, wt_sm, fb + bo2, nullptr, xf, xb, R_, D_, D_, 1);
  }

  // ---- Head: LN2 -> fc2 (N=32 -> 32x32 wave tiles) ----
  ln_kernel<<<R_, 256, 0, stream>>>(xf, ln2g, ln2b, ylnbf, D_);
  wtrans_kernel<<<(D_ * 32) / 256, 256, 0, stream>>>(fc2W, wt_sm, D_, 32);
  {
    int waves = (R_ / 32) * (32 / 32);
    gemm_bf16_kernel<2, 2><<<(waves + 7) / 8, 256, 0, stream>>>(
        ylnbf, wt_sm, fc2b, nullptr, (float*)d_out, nullptr, R_, 32, D_, 0);
  }

  (void)in_sizes; (void)n_in; (void)out_size; (void)ws_size;
}
